// WLSMLPLayerEdge_49065706389982
// MI455X (gfx1250) — compile-verified, matrix operand order checked by reading, not run on hardware
//
#include <hip/hip_runtime.h>

// ---------------------------------------------------------------------------
// CDNA5 bf16 WMMA types
// ---------------------------------------------------------------------------
typedef __attribute__((ext_vector_type(16))) __bf16 v16bf;
typedef __attribute__((ext_vector_type(8)))  float  v8f;

union BFrag {
    v16bf v;
    uint4 q[2];
};

static __device__ __forceinline__ v8f wmma_bf16(const BFrag& a, const BFrag& b, v8f c) {
    return __builtin_amdgcn_wmma_f32_16x16x32_bf16(
        /*neg_a=*/false, a.v, /*neg_b=*/false, b.v,
        /*c_mod=*/(short)0, c, /*reuse_a=*/false, /*reuse_b=*/false);
}

// Wave-local LDS fence: same-wave DS ops are in-order; the wait + memory
// clobber stops the compiler from hoisting cross-lane LDS reads over writes.
#define LDS_FENCE() asm volatile("s_wait_dscnt 0" ::: "memory")

// ---------------------------------------------------------------------------
// Repack f32 row-major weight W[K][N] into bf16 WMMA B-fragments.
// Tile = 32(K) x 16(N). Fragment element order per ISA 7.12.2 (16-bit matrix):
//   lane l: n = nTile*16 + (l&15), half h = l>>4
//   elem e (0..15): j = e>>1, p = e&1
//   k_local = (j<4 ? 2j : 16 + 2(j-4)) + 8h + p
// frag[tile*512 + lane*16 + elem] -> each lane loads one contiguous 32B chunk.
// ---------------------------------------------------------------------------
__global__ __launch_bounds__(256) void pack_weights_kernel(
    const float* __restrict__ W, __bf16* __restrict__ frag, int K, int N)
{
    int g = blockIdx.x * 256 + threadIdx.x;
    if (g >= K * N) return;
    int tile   = g >> 9;
    int within = g & 511;
    int lane   = within >> 4;
    int elem   = within & 15;
    int nTiles = N >> 4;
    int kTile  = tile / nTiles;
    int nTile  = tile - kTile * nTiles;
    int h      = lane >> 4;
    int n      = (nTile << 4) + (lane & 15);
    int j      = elem >> 1;
    int p      = elem & 1;
    int kl     = ((j < 4) ? (j << 1) : (16 + ((j - 4) << 1))) + (h << 3) + p;
    int k      = (kTile << 5) + kl;
    frag[g] = (__bf16)W[k * N + n];
}

__global__ __launch_bounds__(256) void zero_out_kernel(float4* __restrict__ p, int n4)
{
    int i = blockIdx.x * 256 + threadIdx.x;
    if (i < n4) p[i] = make_float4(0.f, 0.f, 0.f, 0.f);
}

// ---------------------------------------------------------------------------
// Edge message kernel: ONE wave per block, 32 edges per wave (two 16-row
// M-tiles sharing every B fragment -> half the L2 weight traffic, two
// independent WMMA accumulator chains).
//   msg_in [32 x 256] = [e_emb[type] | feat[src]]  (bf16 in LDS)
//   hid = relu(msg_in @ W1e + b1e)   [32 x 512]    (bf16 in LDS)
//   m   = hid @ W2e + b2e            [32 x 128]
//   atomicAdd into out[dst][128 + col]
// ---------------------------------------------------------------------------
__global__ __launch_bounds__(32) void edge_msg_kernel(
    const float*  __restrict__ feat,
    const int*    __restrict__ esrc,
    const int*    __restrict__ edst,
    const int*    __restrict__ etype,
    const float*  __restrict__ e_emb,
    const __bf16* __restrict__ w1f, const float* __restrict__ b1,
    const __bf16* __restrict__ w2f, const float* __restrict__ b2,
    float* __restrict__ out, int n_edges)
{
    __shared__ __align__(16) __bf16 s_msg[2][16][256];   // 16 KB
    __shared__ __align__(16) __bf16 s_hid[2][16][512];   // 32 KB
    __shared__ int s_dst[32];

    const int lane = threadIdx.x;
    const int e0   = blockIdx.x * 32;
    if (e0 >= n_edges) return;

    // ---- stage 32 msg_in rows as bf16 (cols 0..127 emb, 128..255 feat)
    for (int idx = lane; idx < 32 * 64; idx += 32) {
        int row = idx >> 6;            // 0..31
        int c4  = (idx & 63) << 2;
        int e   = e0 + row; if (e >= n_edges) e = n_edges - 1;
        const float* src = (c4 < 128) ? (e_emb + etype[e] * 128 + c4)
                                      : (feat  + esrc[e]  * 128 + (c4 - 128));
        float4 v = *(const float4*)src;
        __bf16* d = &s_msg[row >> 4][row & 15][c4];
        d[0] = (__bf16)v.x; d[1] = (__bf16)v.y; d[2] = (__bf16)v.z; d[3] = (__bf16)v.w;
    }
    {
        int e = e0 + lane;
        s_dst[lane] = (e < n_edges) ? edst[e] : -1;
    }
    LDS_FENCE();

    const int h  = lane >> 4;   // K sub-chunk select for A/B fragments
    const int nl = lane & 15;   // M for A-frag rows / N for C-D cols

    // ---- GEMM1: [32x256] @ [256x512] -> relu -> s_hid (bf16)
    for (int nt = 0; nt < 32; ++nt) {
        float bias = b1[nt * 16 + nl];
        v8f acc0, acc1;
        #pragma unroll
        for (int j = 0; j < 8; ++j) { acc0[j] = bias; acc1[j] = bias; }
        #pragma unroll
        for (int kt = 0; kt < 8; ++kt) {
            BFrag a0, a1, b;
            const __bf16* bp = w1f + (size_t)(kt * 32 + nt) * 512 + lane * 16;
            b.q[0] = *(const uint4*)(bp);
            b.q[1] = *(const uint4*)(bp + 8);
            const __bf16* ap0 = &s_msg[0][nl][kt * 32 + h * 8];
            a0.q[0] = *(const uint4*)(ap0);
            a0.q[1] = *(const uint4*)(ap0 + 16);
            const __bf16* ap1 = &s_msg[1][nl][kt * 32 + h * 8];
            a1.q[0] = *(const uint4*)(ap1);
            a1.q[1] = *(const uint4*)(ap1 + 16);
            acc0 = wmma_bf16(a0, b, acc0);
            acc1 = wmma_bf16(a1, b, acc1);
        }
        #pragma unroll
        for (int j = 0; j < 8; ++j) {
            float v0 = acc0[j]; v0 = v0 > 0.f ? v0 : 0.f;
            float v1 = acc1[j]; v1 = v1 > 0.f ? v1 : 0.f;
            s_hid[0][j + h * 8][nt * 16 + nl] = (__bf16)v0;
            s_hid[1][j + h * 8][nt * 16 + nl] = (__bf16)v1;
        }
    }
    LDS_FENCE();

    // ---- GEMM2: [32x512] @ [512x128] + b2 -> atomic scatter
    for (int nt = 0; nt < 8; ++nt) {
        float bias = b2[nt * 16 + nl];
        v8f acc0, acc1;
        #pragma unroll
        for (int j = 0; j < 8; ++j) { acc0[j] = bias; acc1[j] = bias; }
        #pragma unroll
        for (int kt = 0; kt < 16; ++kt) {
            BFrag a0, a1, b;
            const __bf16* bp = w2f + (size_t)(kt * 8 + nt) * 512 + lane * 16;
            b.q[0] = *(const uint4*)(bp);
            b.q[1] = *(const uint4*)(bp + 8);
            const __bf16* ap0 = &s_hid[0][nl][kt * 32 + h * 8];
            a0.q[0] = *(const uint4*)(ap0);
            a0.q[1] = *(const uint4*)(ap0 + 16);
            const __bf16* ap1 = &s_hid[1][nl][kt * 32 + h * 8];
            a1.q[0] = *(const uint4*)(ap1);
            a1.q[1] = *(const uint4*)(ap1 + 16);
            acc0 = wmma_bf16(a0, b, acc0);
            acc1 = wmma_bf16(a1, b, acc1);
        }
        #pragma unroll
        for (int j = 0; j < 8; ++j) {
            int d0 = s_dst[j + h * 8];
            int d1 = s_dst[16 + j + h * 8];
            if (d0 >= 0) atomicAdd(out + (size_t)d0 * 256 + 128 + nt * 16 + nl, acc0[j]);
            if (d1 >= 0) atomicAdd(out + (size_t)d1 * 256 + 128 + nt * 16 + nl, acc1[j]);
        }
    }
}

// ---------------------------------------------------------------------------
// Self-MLP kernel: one wave per block, 32 nodes per wave (two M-tiles).
//   hid = relu(feat @ W1s + b1s)  [32 x 256]
//   o   = hid @ W2s + b2s         [32 x 128] -> out[:, 0:128] plain stores
// ---------------------------------------------------------------------------
__global__ __launch_bounds__(32) void self_mlp_kernel(
    const float*  __restrict__ feat,
    const __bf16* __restrict__ w1f, const float* __restrict__ b1,
    const __bf16* __restrict__ w2f, const float* __restrict__ b2,
    float* __restrict__ out, int n_nodes)
{
    __shared__ __align__(16) __bf16 s_in[2][16][128];    // 8 KB
    __shared__ __align__(16) __bf16 s_hid[2][16][256];   // 16 KB

    const int lane = threadIdx.x;
    const int n0   = blockIdx.x * 32;
    if (n0 >= n_nodes) return;

    for (int idx = lane; idx < 32 * 32; idx += 32) {
        int row = idx >> 5;            // 0..31
        int c4  = (idx & 31) << 2;
        int n   = n0 + row; if (n >= n_nodes) n = n_nodes - 1;
        float4 v = *(const float4*)(feat + (size_t)n * 128 + c4);
        __bf16* d = &s_in[row >> 4][row & 15][c4];
        d[0] = (__bf16)v.x; d[1] = (__bf16)v.y; d[2] = (__bf16)v.z; d[3] = (__bf16)v.w;
    }
    LDS_FENCE();

    const int h  = lane >> 4;
    const int nl = lane & 15;

    // GEMM1: [32x128] @ [128x256] -> relu
    for (int nt = 0; nt < 16; ++nt) {
        float bias = b1[nt * 16 + nl];
        v8f acc0, acc1;
        #pragma unroll
        for (int j = 0; j < 8; ++j) { acc0[j] = bias; acc1[j] = bias; }
        #pragma unroll
        for (int kt = 0; kt < 4; ++kt) {
            BFrag a0, a1, b;
            const __bf16* bp = w1f + (size_t)(kt * 16 + nt) * 512 + lane * 16;
            b.q[0] = *(const uint4*)(bp);
            b.q[1] = *(const uint4*)(bp + 8);
            const __bf16* ap0 = &s_in[0][nl][kt * 32 + h * 8];
            a0.q[0] = *(const uint4*)(ap0);
            a0.q[1] = *(const uint4*)(ap0 + 16);
            const __bf16* ap1 = &s_in[1][nl][kt * 32 + h * 8];
            a1.q[0] = *(const uint4*)(ap1);
            a1.q[1] = *(const uint4*)(ap1 + 16);
            acc0 = wmma_bf16(a0, b, acc0);
            acc1 = wmma_bf16(a1, b, acc1);
        }
        #pragma unroll
        for (int j = 0; j < 8; ++j) {
            float v0 = acc0[j]; v0 = v0 > 0.f ? v0 : 0.f;
            float v1 = acc1[j]; v1 = v1 > 0.f ? v1 : 0.f;
            s_hid[0][j + h * 8][nt * 16 + nl] = (__bf16)v0;
            s_hid[1][j + h * 8][nt * 16 + nl] = (__bf16)v1;
        }
    }
    LDS_FENCE();

    // GEMM2: [32x256] @ [256x128] + b2 -> out[:, 0:128]
    for (int nt = 0; nt < 8; ++nt) {
        float bias = b2[nt * 16 + nl];
        v8f acc0, acc1;
        #pragma unroll
        for (int j = 0; j < 8; ++j) { acc0[j] = bias; acc1[j] = bias; }
        #pragma unroll
        for (int kt = 0; kt < 8; ++kt) {
            BFrag a0, a1, b;
            const __bf16* bp = w2f + (size_t)(kt * 8 + nt) * 512 + lane * 16;
            b.q[0] = *(const uint4*)(bp);
            b.q[1] = *(const uint4*)(bp + 8);
            const __bf16* ap0 = &s_hid[0][nl][kt * 32 + h * 8];
            a0.q[0] = *(const uint4*)(ap0);
            a0.q[1] = *(const uint4*)(ap0 + 16);
            const __bf16* ap1 = &s_hid[1][nl][kt * 32 + h * 8];
            a1.q[0] = *(const uint4*)(ap1);
            a1.q[1] = *(const uint4*)(ap1 + 16);
            acc0 = wmma_bf16(a0, b, acc0);
            acc1 = wmma_bf16(a1, b, acc1);
        }
        #pragma unroll
        for (int j = 0; j < 8; ++j) {
            int na = n0 + j + h * 8;
            int nb = n0 + 16 + j + h * 8;
            if (na < n_nodes) out[(size_t)na * 256 + nt * 16 + nl] = acc0[j];
            if (nb < n_nodes) out[(size_t)nb * 256 + nt * 16 + nl] = acc1[j];
        }
    }
}

// ---------------------------------------------------------------------------
// Launch
// ---------------------------------------------------------------------------
extern "C" void kernel_launch(void* const* d_in, const int* in_sizes, int n_in,
                              void* d_out, int out_size, void* d_ws, size_t ws_size,
                              hipStream_t stream)
{
    const float* feat  = (const float*)d_in[0];
    const int*   esrc  = (const int*)  d_in[1];
    const int*   edst  = (const int*)  d_in[2];
    const int*   etype = (const int*)  d_in[3];
    const float* e_emb = (const float*)d_in[4];
    const float* W1e   = (const float*)d_in[5];
    const float* b1e   = (const float*)d_in[6];
    const float* W2e   = (const float*)d_in[7];
    const float* b2e   = (const float*)d_in[8];
    const float* W1s   = (const float*)d_in[9];
    const float* b1s   = (const float*)d_in[10];
    const float* W2s   = (const float*)d_in[11];
    const float* b2s   = (const float*)d_in[12];
    float* out = (float*)d_out;

    const int n_nodes = in_sizes[0] / 128;
    const int n_edges = in_sizes[1];

    // workspace layout: bf16 weight fragments (512 KB total)
    __bf16* w1e_f = (__bf16*)d_ws;              // 256*512
    __bf16* w2e_f = w1e_f + 256 * 512;          // 512*128
    __bf16* w1s_f = w2e_f + 512 * 128;          // 128*256
    __bf16* w2s_f = w1s_f + 128 * 256;          // 256*128

    // zero output (agg half is accumulated with atomics every call)
    int n4 = out_size / 4;
    zero_out_kernel<<<(n4 + 255) / 256, 256, 0, stream>>>((float4*)d_out, n4);

    pack_weights_kernel<<<(256 * 512 + 255) / 256, 256, 0, stream>>>(W1e, w1e_f, 256, 512);
    pack_weights_kernel<<<(512 * 128 + 255) / 256, 256, 0, stream>>>(W2e, w2e_f, 512, 128);
    pack_weights_kernel<<<(128 * 256 + 255) / 256, 256, 0, stream>>>(W1s, w1s_f, 128, 256);
    pack_weights_kernel<<<(256 * 128 + 255) / 256, 256, 0, stream>>>(W2s, w2s_f, 256, 128);

    int nblocks = (n_nodes + 31) / 32;
    self_mlp_kernel<<<nblocks, 32, 0, stream>>>(
        feat, w1s_f, b1s, w2s_f, b2s, out, n_nodes);

    int eblocks = (n_edges + 31) / 32;
    edge_msg_kernel<<<eblocks, 32, 0, stream>>>(
        feat, esrc, edst, etype, e_emb, w1e_f, b1e, w2e_f, b2e, out, n_edges);
}
